// AttentionLayer_76398878261268
// MI455X (gfx1250) — matrix operand order repaired and checked by reading, not compiled, not run on hardware
//
#include <hip/hip_runtime.h>
#include <cstdint>
#include <cstddef>

// Problem constants
#define BB 16
#define NN 2048
#define DD 512

#define CHUNK_ROWS 32                 // 2 sixteen-row tiles per staged chunk
#define NCHUNK (NN / CHUNK_ROWS)      // 64 chunks over the streamed dimension
#define CHUNK_BYTES (CHUNK_ROWS * DD * 2)

typedef __attribute__((ext_vector_type(16))) __bf16 v16bf;
typedef __attribute__((ext_vector_type(8)))  float  v8f;

union Frag {
    v16bf    v;
    uint32_t u[8];
};

__device__ __forceinline__ float bf2f(uint16_t h) {
    union { uint32_t u; float f; } c; c.u = ((uint32_t)h) << 16; return c.f;
}
__device__ __forceinline__ uint16_t f2bf(float f) {
    union { float f; uint32_t u; } c{f};
    uint32_t r = c.u + 0x7FFFu + ((c.u >> 16) & 1u);  // round-to-nearest-even
    return (uint16_t)(r >> 16);
}

// A-fragment K index for VGPR vv (0..7), lane-half h: pairs (K, K+1)
__device__ __forceinline__ int a_kidx(int vv, int h) {
    return 2 * (vv & 3) + 16 * (vv >> 2) + 8 * h;
}

// Issue one async 16-byte global->LDS copy (GVS mode: 64b saddr + 32b voffset).
__device__ __forceinline__ void async_copy_b128(const void* lds_ptr,
                                                uint32_t voff, uint64_t sbase) {
    uint32_t lds_addr = (uint32_t)(uintptr_t)lds_ptr;  // LDS address is the pointer's low 32 bits
    asm volatile("global_load_async_to_lds_b128 %0, %1, %2"
                 :: "v"(lds_addr), "v"(voff), "s"(sbase) : "memory");
}
__device__ __forceinline__ void wait_async_le8() {
    asm volatile("s_wait_asynccnt 0x8" ::: "memory");
}
__device__ __forceinline__ void wait_async_0() {
    asm volatile("s_wait_asynccnt 0x0" ::: "memory");
}

// ---------------------------------------------------------------------------
// Kernel 0: convert + transpose the three weight matrices to bf16.
// Wt[w][col][kk] = W_w[kk][col]   (so B-fragment K-pairs are contiguous)
// ---------------------------------------------------------------------------
__global__ __launch_bounds__(512) void wtrans_kernel(
    const float* __restrict__ Wk, const float* __restrict__ Wq,
    const float* __restrict__ Wv, uint16_t* __restrict__ Wt) {
    int kk  = blockIdx.x;   // 0..511
    int w   = blockIdx.y;   // 0..2
    int col = threadIdx.x;  // 0..511
    const float* src = (w == 0) ? Wk : ((w == 1) ? Wq : Wv);
    Wt[(size_t)w * DD * DD + (size_t)col * DD + kk] = f2bf(src[(size_t)kk * DD + col]);
}

// ---------------------------------------------------------------------------
// Kernel 1: projections  k = x@Wk, q = x@Wq, v = x@Wv  (bf16 out, f32 acc)
// ---------------------------------------------------------------------------
__global__ __launch_bounds__(256) void proj_kernel(
    const float* __restrict__ x, const uint16_t* __restrict__ Wt,
    uint16_t* __restrict__ kb, uint16_t* __restrict__ qb, uint16_t* __restrict__ vb) {
    __shared__ uint16_t xs[16 * DD];  // 16 KB bf16 stripe

    const int tid = threadIdx.x;
    for (int i = tid; i < 16 * DD; i += 256) {
        int row = i >> 9, d = i & (DD - 1);
        xs[i] = f2bf(x[((size_t)blockIdx.x * 16 + row) * DD + d]);
    }
    __syncthreads();

    const int wave = tid >> 5, lane = tid & 31;
    const int m = lane & 15, h = lane >> 4;

    v8f acc[12] = {};  // [weight 0..2][coltile 0..3]

    for (int ks = 0; ks < 16; ++ks) {
        const int kbase = ks * 32;
        Frag a;
#pragma unroll
        for (int vv = 0; vv < 8; ++vv)
            a.u[vv] = *(const uint32_t*)&xs[m * DD + kbase + a_kidx(vv, h)];

#pragma unroll
        for (int t = 0; t < 12; ++t) {
            const int widx  = t >> 2;
            const int ntile = wave * 4 + (t & 3);
            const int col   = ntile * 16 + m;
            const uint16_t* wt = Wt + (size_t)widx * DD * DD + (size_t)col * DD;
            const int kst = kbase + 16 * h;
            Frag b;
#pragma unroll
            for (int vv = 0; vv < 8; ++vv)
                b.u[vv] = *(const uint32_t*)&wt[kbase + a_kidx(vv, h)];
            acc[t] = __builtin_amdgcn_wmma_f32_16x16x32_bf16(
                false, a.v, false, b.v, (short)0, acc[t], false, false);
        }
    }

    uint16_t* outs[3] = {kb, qb, vb};
#pragma unroll
    for (int t = 0; t < 12; ++t) {
        const int widx  = t >> 2;
        const int ntile = wave * 4 + (t & 3);
        const int col   = ntile * 16 + m;
#pragma unroll
        for (int r = 0; r < 8; ++r) {
            size_t row = (size_t)blockIdx.x * 16 + r + 8 * h;
            outs[widx][row * DD + col] = f2bf(acc[t][r]);
        }
    }
}

// ---------------------------------------------------------------------------
// Kernel 2: per-query-row softmax stats (running max, sum-exp) over all keys.
// 8 waves per block share one batch; K stream staged cooperatively in LDS via
// double-buffered async copies. Q fragments stay register-resident per wave.
// ---------------------------------------------------------------------------
__global__ __launch_bounds__(256) void stats_kernel(
    const uint16_t* __restrict__ qb, const uint16_t* __restrict__ kb,
    float* __restrict__ stats) {
    __shared__ uint16_t sK[2][CHUNK_ROWS * DD];  // 2 x 32 KB

    const int tid  = threadIdx.x;
    const int wave = tid >> 5, lane = tid & 31;
    const int m = lane & 15, h = lane >> 4;
    const int b  = blockIdx.x >> 4;                  // 16 blocks per batch
    const int qt = (blockIdx.x & 15) * 8 + wave;     // this wave's q-tile
    const float scale = 0.0441941738241592f;         // 1/sqrt(512)

    const uint64_t kbase = (uint64_t)(uintptr_t)(kb + (size_t)b * NN * DD);

    // preload Q fragments for all 16 k-steps (register-resident)
    const uint16_t* qrow = qb + ((size_t)b * NN + qt * 16 + m) * DD;
    Frag qf[16];
#pragma unroll
    for (int ks = 0; ks < 16; ++ks)
#pragma unroll
        for (int vv = 0; vv < 8; ++vv)
            qf[ks].u[vv] = *(const uint32_t*)&qrow[ks * 32 + a_kidx(vv, h)];

    float rmax[8], rsum[8];
#pragma unroll
    for (int r = 0; r < 8; ++r) { rmax[r] = -3.0e38f; rsum[r] = 0.0f; }

    // stage chunk 0
#pragma unroll
    for (int i = 0; i < 8; ++i) {
        int e = tid + i * 256;  // 0..2047 sixteen-byte packets
        async_copy_b128(&sK[0][e * 8], (uint32_t)(e * 16), kbase);
    }

    for (int ch = 0; ch < NCHUNK; ++ch) {
        const int cur = ch & 1;
        if (ch + 1 < NCHUNK) {
#pragma unroll
            for (int i = 0; i < 8; ++i) {
                int e = tid + i * 256;
                async_copy_b128(&sK[cur ^ 1][e * 8],
                                (uint32_t)((ch + 1) * CHUNK_BYTES + e * 16), kbase);
            }
            wait_async_le8();   // previous chunk landed; next chunk in flight
        } else {
            wait_async_0();
        }
        __syncthreads();

#pragma unroll
        for (int t2 = 0; t2 < 2; ++t2) {
            const uint16_t* krow = &sK[cur][(t2 * 16 + m) * DD];
            v8f acc = {};
#pragma unroll
            for (int ks = 0; ks < 16; ++ks) {
                const int kst = ks * 32 + 16 * h;
                Frag kf;
#pragma unroll
                for (int vv = 0; vv < 8; ++vv)
                    kf.u[vv] = *(const uint32_t*)&krow[ks * 32 + a_kidx(vv, h)];
                acc = __builtin_amdgcn_wmma_f32_16x16x32_bf16(
                    false, qf[ks].v, false, kf.v, (short)0, acc, false, false);
            }
            // C layout: VGPR r holds row (r + 8h), key column (lane&15)
#pragma unroll
            for (int r = 0; r < 8; ++r) {
                float s = acc[r] * scale;
                float tmax = s;
#pragma unroll
                for (int off = 1; off < 16; off <<= 1)
                    tmax = fmaxf(tmax, __shfl_xor(tmax, off, 16));
                float nm = fmaxf(rmax[r], tmax);
                float p  = __expf(s - nm);
                float ts = p;
#pragma unroll
                for (int off = 1; off < 16; off <<= 1)
                    ts += __shfl_xor(ts, off, 16);
                rsum[r] = rsum[r] * __expf(rmax[r] - nm) + ts;
                rmax[r] = nm;
            }
        }
        __syncthreads();
    }

    if (m == 0) {  // lane 0 writes rows 0..7, lane 16 writes rows 8..15
#pragma unroll
        for (int r = 0; r < 8; ++r) {
            size_t row = (size_t)b * NN + qt * 16 + r + 8 * h;
            stats[row * 2 + 0] = rmax[r];
            stats[row * 2 + 1] = rsum[r];
        }
    }
}

// ---------------------------------------------------------------------------
// Kernel 3: column sums of the attention-weight matrix:
//   c[b,m] = sum_n exp(s[n,m]-max_n)/l_n
// 8 waves per block share one batch; Q stream staged in LDS via async copies.
// K fragments stay register-resident per wave.
// ---------------------------------------------------------------------------
__global__ __launch_bounds__(256) void colsum_kernel(
    const uint16_t* __restrict__ qb, const uint16_t* __restrict__ kb,
    const float* __restrict__ stats, float* __restrict__ cout) {
    __shared__ uint16_t sQ[2][CHUNK_ROWS * DD];  // 2 x 32 KB

    const int tid  = threadIdx.x;
    const int wave = tid >> 5, lane = tid & 31;
    const int m = lane & 15, h = lane >> 4;
    const int b  = blockIdx.x >> 4;
    const int kt = (blockIdx.x & 15) * 8 + wave;     // this wave's key-tile
    const float scale = 0.0441941738241592f;

    const uint64_t qbase = (uint64_t)(uintptr_t)(qb + (size_t)b * NN * DD);

    // preload K fragments (B operand) for this key tile
    const uint16_t* krow = kb + ((size_t)b * NN + kt * 16 + m) * DD;
    Frag kf[16];
#pragma unroll
    for (int ks = 0; ks < 16; ++ks)
#pragma unroll
        for (int vv = 0; vv < 8; ++vv)
            kf[ks].u[vv] = *(const uint32_t*)&krow[ks * 32 + a_kidx(vv, h)];

    float cacc = 0.0f;

#pragma unroll
    for (int i = 0; i < 8; ++i) {
        int e = tid + i * 256;
        async_copy_b128(&sQ[0][e * 8], (uint32_t)(e * 16), qbase);
    }

    for (int ch = 0; ch < NCHUNK; ++ch) {
        const int cur = ch & 1;
        if (ch + 1 < NCHUNK) {
#pragma unroll
            for (int i = 0; i < 8; ++i) {
                int e = tid + i * 256;
                async_copy_b128(&sQ[cur ^ 1][e * 8],
                                (uint32_t)((ch + 1) * CHUNK_BYTES + e * 16), qbase);
            }
            wait_async_le8();
        } else {
            wait_async_0();
        }
        __syncthreads();

#pragma unroll
        for (int t2 = 0; t2 < 2; ++t2) {
            const int qt = ch * 2 + t2;
            const uint16_t* qrowl = &sQ[cur][(t2 * 16 + m) * DD];
            v8f acc = {};
#pragma unroll
            for (int ks = 0; ks < 16; ++ks) {
                Frag af;
#pragma unroll
                for (int vv = 0; vv < 8; ++vv)
                    af.u[vv] = *(const uint32_t*)&qrowl[ks * 32 + a_kidx(vv, h)];
                acc = __builtin_amdgcn_wmma_f32_16x16x32_bf16(
                    false, af.v, false, kf[ks].v, (short)0, acc, false, false);
            }
#pragma unroll
            for (int r = 0; r < 8; ++r) {
                size_t row = (size_t)b * NN + qt * 16 + r + 8 * h;
                float mx = stats[row * 2 + 0];
                float sm = stats[row * 2 + 1];
                cacc += __expf(acc[r] * scale - mx) / sm;
            }
        }
        __syncthreads();
    }

    cacc += __shfl_xor(cacc, 16, 32);  // fold the two row halves
    if (lane < 16)
        cout[(size_t)b * NN + kt * 16 + lane] = cacc;
}

// ---------------------------------------------------------------------------
// Kernel 4: out[b,d] = (1/N) * sum_m c[b,m] * v[b,m,d]   (tiny)
// ---------------------------------------------------------------------------
__global__ __launch_bounds__(256) void out_kernel(
    const float* __restrict__ c, const uint16_t* __restrict__ vb,
    float* __restrict__ out) {
    const int b = blockIdx.x >> 1;
    const int d = (blockIdx.x & 1) * 256 + threadIdx.x;
    float acc = 0.0f;
    for (int mkey = 0; mkey < NN; ++mkey)
        acc += c[(size_t)b * NN + mkey] * bf2f(vb[((size_t)b * NN + mkey) * DD + d]);
    out[(size_t)b * DD + d] = acc * (1.0f / NN);
}

// ---------------------------------------------------------------------------
extern "C" void kernel_launch(void* const* d_in, const int* in_sizes, int n_in,
                              void* d_out, int out_size, void* d_ws, size_t ws_size,
                              hipStream_t stream) {
    const float* x  = (const float*)d_in[0];
    const float* Wk = (const float*)d_in[1];
    const float* Wq = (const float*)d_in[2];
    const float* Wv = (const float*)d_in[3];
    float* out = (float*)d_out;

    char* p = (char*)d_ws;
    uint16_t* kb = (uint16_t*)p;  p += (size_t)BB * NN * DD * 2;
    uint16_t* qb = (uint16_t*)p;  p += (size_t)BB * NN * DD * 2;
    uint16_t* vb = (uint16_t*)p;  p += (size_t)BB * NN * DD * 2;
    uint16_t* Wt = (uint16_t*)p;  p += (size_t)3 * DD * DD * 2;
    float* stats = (float*)p;     p += (size_t)BB * NN * 2 * sizeof(float);
    float* cvec  = (float*)p;

    wtrans_kernel<<<dim3(DD, 3), 512, 0, stream>>>(Wk, Wq, Wv, Wt);
    proj_kernel<<<dim3((BB * NN) / 16), 256, 0, stream>>>(x, Wt, kb, qb, vb);
    stats_kernel<<<dim3(256), 256, 0, stream>>>(qb, kb, stats);
    colsum_kernel<<<dim3(256), 256, 0, stream>>>(qb, kb, stats, cvec);
    out_kernel<<<dim3(BB * 2), 256, 0, stream>>>(cvec, vb, out);
}
